// FlowEmbedding_48163763257800
// MI455X (gfx1250) — matrix operand order repaired and checked by reading, not compiled, not run on hardware
//
#include <hip/hip_runtime.h>
#include <hip/hip_bf16.h>

typedef __attribute__((ext_vector_type(16))) __bf16 v16bf;
typedef __attribute__((ext_vector_type(8)))  __bf16 v8bf;
typedef __attribute__((ext_vector_type(8)))  float  v8f;

#define B_   4
#define N_   2048
#define K_   32
#define C_   128
#define M_   (B_*N_*K_)   // 262144
#define CIN0 259
#define CP0  288          // 259 padded to multiple of 32
#define CP0S 296          // LDS row stride (bf16 elems), keeps 16B alignment
#define CP1S 136
#define MT   64           // m-positions per block (= 2 n-points x K)

// ---------------- prep: weights -> bf16 (padded), zero stats, copy pos1 ----
__global__ void prep_kernel(const float* __restrict__ W0, const float* __restrict__ W1,
                            const float* __restrict__ W2, const float* __restrict__ pos1,
                            __bf16* __restrict__ wb0, __bf16* __restrict__ wb1,
                            __bf16* __restrict__ wb2, float* __restrict__ stats,
                            float* __restrict__ out) {
  int i0 = blockIdx.x * blockDim.x + threadIdx.x;
  int stride = gridDim.x * blockDim.x;
  for (int i = i0; i < 128*CP0; i += stride) {
    int o = i / CP0, c = i % CP0;
    wb0[i] = (c < CIN0) ? (__bf16)W0[o*CIN0 + c] : (__bf16)0.f;
  }
  for (int i = i0; i < 128*128; i += stride) wb1[i] = (__bf16)W1[i];
  for (int i = i0; i < 256*128; i += stride) wb2[i] = (__bf16)W2[i];
  for (int i = i0; i < 1024;    i += stride) stats[i] = 0.f;
  for (int i = i0; i < B_*3*N_; i += stride) out[i] = pos1[i];
}

// Cross-lane (16-lane half) reduce of register-accumulated per-channel sums,
// committed straight to the global stat accumulators. Channels are disjoint
// across waves and across the two lane-halves, so exactly one atomic per
// channel per block. Executed ONCE per wave per out-tile group.
static __device__ __forceinline__ void stat_commit(const float* ssum, const float* ssq,
                                                   int co, int lrow,
                                                   float* __restrict__ gsum,
                                                   float* __restrict__ gsq) {
#pragma unroll
  for (int r = 0; r < 8; ++r) {
    float v = ssum[r];
    float q = ssq[r];
#pragma unroll
    for (int off = 1; off < 16; off <<= 1) {
      v += __shfl_xor(v, off, 32);
      q += __shfl_xor(q, off, 32);
    }
    if (lrow == 0) {
      atomicAdd(&gsum[co + r], v);
      atomicAdd(&gsq[co + r],  q);
    }
  }
}

// ---------------- layer 0: gather+concat -> WMMA GEMM (288->128) -----------
__global__ void __launch_bounds__(256) layer0_kernel(
    const float* __restrict__ pos1, const float* __restrict__ pos2,
    const float* __restrict__ feat1, const float* __restrict__ feat2,
    const int* __restrict__ idx, const __bf16* __restrict__ wb0,
    __bf16* __restrict__ y0, float* __restrict__ stats0) {
  __shared__ __align__(16) __bf16 xs[MT*CP0S];
  __shared__ int s_idx[MT];
  int tid = threadIdx.x;
  int m0 = blockIdx.x * MT;
  int b  = m0 >> 16;                       // batch: uniform per block
  if (tid < MT) s_idx[tid] = idx[m0 + tid];
  __syncthreads();

  // feat2 gather: channels 3..130 ; lanes walk m (random 4B in one 8KB row)
  {
    const float* f2b = feat2 + (size_t)b * C_ * N_;
    for (int e = tid; e < MT*128; e += 256) {
      int ml = e & 63, cc = e >> 6;
      xs[ml*CP0S + 3 + cc] = (__bf16)f2b[cc*N_ + s_idx[ml]];
    }
  }
  // pos_diff: channels 0..2
  {
    int ml = tid >> 2, c = tid & 3;
    if (c < 3) {
      int n = ((m0 + ml) >> 5) & (N_ - 1);
      xs[ml*CP0S + c] =
          (__bf16)(pos2[(b*3 + c)*N_ + s_idx[ml]] - pos1[(b*3 + c)*N_ + n]);
    }
  }
  // zero pad: channels 259..287
  for (int e = tid; e < MT*32; e += 256) {
    int ml = e >> 5, c = 256 + (e & 31);
    if (c >= CIN0) xs[ml*CP0S + c] = (__bf16)0.f;
  }
  // feat1 broadcast: channels 131..258 ; only 2 distinct n per block
  {
    int j = tid >> 7, c = tid & 127;         // j = n-group (0/1)
    int n = ((m0 + j*K_) >> 5) & (N_ - 1);
    __bf16 bv = (__bf16)feat1[(b*C_ + c)*N_ + n];
#pragma unroll
    for (int k = 0; k < K_; ++k) xs[(j*K_ + k)*CP0S + 131 + c] = bv;
  }
  __syncthreads();

  int wave = tid >> 5, lane = tid & 31;
  int lrow = lane & 15, lhi = lane >> 4;
  int o0 = wave * 16;                      // each wave owns one out-tile
  int co = o0 + lhi*8;

  // A (weight) tiles live in registers across the m-subtile loop
  v16bf amat[9];
  {
    const __bf16* abase = wb0 + (o0 + lrow)*CP0 + lhi*8;
#pragma unroll
    for (int i = 0; i < 9; ++i) {
      ((v8bf*)&amat[i])[0] = *(const v8bf*)(abase + i*32);
      ((v8bf*)&amat[i])[1] = *(const v8bf*)(abase + i*32 + 16);
    }
  }
  float ssum[8] = {0,0,0,0,0,0,0,0};
  float ssq[8]  = {0,0,0,0,0,0,0,0};
#pragma unroll
  for (int ms4 = 0; ms4 < 4; ++ms4) {      // iterate the 4 m-subtiles
    int ms = ms4 * 16;
    const __bf16* bbase = xs + (ms + lrow)*CP0S + lhi*8;
    v8f acc = {};
#pragma unroll
    for (int i = 0; i < 9; ++i) {
      v16bf bm;
      ((v8bf*)&bm)[0] = *(const v8bf*)(bbase + i*32);
      ((v8bf*)&bm)[1] = *(const v8bf*)(bbase + i*32 + 16);
      acc = __builtin_amdgcn_wmma_f32_16x16x32_bf16(false, amat[i], false, bm,
                                                    (short)0, acc, false, false);
    }
    int mg = m0 + ms + lrow;
    v8bf ov;
#pragma unroll
    for (int r = 0; r < 8; ++r) {
      ov[r] = (__bf16)acc[r];
      ssum[r] += acc[r];
      ssq[r]   = fmaf(acc[r], acc[r], ssq[r]);
    }
    *(v8bf*)(y0 + (size_t)mg*128 + co) = ov;
  }
  stat_commit(ssum, ssq, co, lrow, stats0, stats0 + 128);
}

// ------------- layers 1/2: BN+LReLU on input tile -> WMMA GEMM (128->COUT) -
__global__ void __launch_bounds__(256) layerN_kernel(
    const __bf16* __restrict__ y_in, const __bf16* __restrict__ wb,
    const float* __restrict__ stats_in, const float* __restrict__ g_in,
    const float* __restrict__ b_in, __bf16* __restrict__ y_out,
    float* __restrict__ stats_out, int COUT) {
  __shared__ __align__(16) __bf16 xs[MT*CP1S];
  __shared__ float s_sc[128], s_bi[128];
  int tid = threadIdx.x;
  const float invM = 1.0f / (float)M_;
  if (tid < 128) {
    float mean = stats_in[tid] * invM;
    float var  = fmaxf(stats_in[128 + tid] * invM - mean*mean, 0.f);
    float sc   = g_in[tid] * rsqrtf(var + 1e-5f);
    s_sc[tid] = sc;
    s_bi[tid] = b_in[tid] - mean*sc;
  }
  __syncthreads();
  int m0 = blockIdx.x * MT;
  // vectorized staging: 8 channels (16B) per thread-step, BN+LReLU fused
  for (int e = tid; e < MT*16; e += 256) {
    int ml = e >> 4, c8 = (e & 15) * 8;
    v8bf yv = *(const v8bf*)(y_in + (size_t)(m0 + ml)*128 + c8);
    v8bf xv;
#pragma unroll
    for (int r = 0; r < 8; ++r) {
      float v = (float)yv[r];
      v = v*s_sc[c8 + r] + s_bi[c8 + r];
      v = v >= 0.f ? v : 0.01f*v;
      xv[r] = (__bf16)v;
    }
    *(v8bf*)(xs + ml*CP1S + c8) = xv;
  }
  __syncthreads();
  int wave = tid >> 5, lane = tid & 31;
  int lrow = lane & 15, lhi = lane >> 4;
  int NOG = COUT >> 7;                     // out-tile groups per wave (1 or 2)

  for (int og = 0; og < NOG; ++og) {
    int o0 = (wave + og*8) * 16;
    int co = o0 + lhi*8;
    v16bf amat[4];
    {
      const __bf16* abase = wb + (o0 + lrow)*128 + lhi*8;
#pragma unroll
      for (int i = 0; i < 4; ++i) {
        ((v8bf*)&amat[i])[0] = *(const v8bf*)(abase + i*32);
        ((v8bf*)&amat[i])[1] = *(const v8bf*)(abase + i*32 + 16);
      }
    }
    float ssum[8] = {0,0,0,0,0,0,0,0};
    float ssq[8]  = {0,0,0,0,0,0,0,0};
#pragma unroll
    for (int ms4 = 0; ms4 < 4; ++ms4) {
      int ms = ms4 * 16;
      const __bf16* bbase = xs + (ms + lrow)*CP1S + lhi*8;
      v8f acc = {};
#pragma unroll
      for (int i = 0; i < 4; ++i) {
        v16bf bm;
        ((v8bf*)&bm)[0] = *(const v8bf*)(bbase + i*32);
        ((v8bf*)&bm)[1] = *(const v8bf*)(bbase + i*32 + 16);
        acc = __builtin_amdgcn_wmma_f32_16x16x32_bf16(false, amat[i], false, bm,
                                                      (short)0, acc, false, false);
      }
      int mg = m0 + ms + lrow;
      v8bf ov;
#pragma unroll
      for (int r = 0; r < 8; ++r) {
        ov[r] = (__bf16)acc[r];
        ssum[r] += acc[r];
        ssq[r]   = fmaf(acc[r], acc[r], ssq[r]);
      }
      *(v8bf*)(y_out + (size_t)mg*COUT + co) = ov;
    }
    stat_commit(ssum, ssq, co, lrow, stats_out, stats_out + COUT);
  }
}

// ---------------- finalize: BN+LReLU on y2 then max over K -----------------
__global__ void __launch_bounds__(256) finalize_kernel(
    const __bf16* __restrict__ y2, const float* __restrict__ stats2,
    const float* __restrict__ g2, const float* __restrict__ b2,
    float* __restrict__ out_feat) {
  int c  = threadIdx.x;     // 0..255 channel
  int bn = blockIdx.x;      // 0..B*N-1
  const float invM = 1.0f / (float)M_;
  float mean = stats2[c] * invM;
  float var  = fmaxf(stats2[256 + c] * invM - mean*mean, 0.f);
  float sc   = g2[c] * rsqrtf(var + 1e-5f);
  float bi   = b2[c] - mean*sc;
  size_t base = (size_t)bn * K_ * 256 + c;
  float mx = -3.4e38f;
#pragma unroll 4
  for (int k = 0; k < K_; ++k) {
    float v = (float)y2[base + (size_t)k*256];
    v = v*sc + bi;
    v = v >= 0.f ? v : 0.01f*v;
    mx = fmaxf(mx, v);
  }
  int b = bn >> 11, n = bn & (N_ - 1);
  out_feat[((size_t)b*256 + c)*N_ + n] = mx;
}

extern "C" void kernel_launch(void* const* d_in, const int* in_sizes, int n_in,
                              void* d_out, int out_size, void* d_ws, size_t ws_size,
                              hipStream_t stream) {
  const float* pos1 = (const float*)d_in[0];
  const float* pos2 = (const float*)d_in[1];
  const float* f1   = (const float*)d_in[2];
  const float* f2   = (const float*)d_in[3];
  const int*   idx  = (const int*)d_in[4];
  const float* W0 = (const float*)d_in[5];
  const float* g0 = (const float*)d_in[6];
  const float* b0 = (const float*)d_in[7];
  const float* W1 = (const float*)d_in[8];
  const float* g1 = (const float*)d_in[9];
  const float* b1 = (const float*)d_in[10];
  const float* W2 = (const float*)d_in[11];
  const float* g2 = (const float*)d_in[12];
  const float* b2 = (const float*)d_in[13];

  char* ws = (char*)d_ws;
  __bf16* wb0   = (__bf16*)(ws);                         // 128*288 bf16
  __bf16* wb1   = (__bf16*)(ws + 73728);                 // 128*128 bf16
  __bf16* wb2   = (__bf16*)(ws + 106496);                // 256*128 bf16
  float*  stats = (float*) (ws + 172032);                // 1024 f32
  __bf16* y0    = (__bf16*)(ws + 176128);                // M*128 bf16
  __bf16* y1    = (__bf16*)(ws + 176128 + 67108864ull);
  __bf16* y2    = (__bf16*)(ws + 176128 + 2ull*67108864ull);
  float* out      = (float*)d_out;
  float* out_feat = out + B_*3*N_;

  prep_kernel  <<<256, 256, 0, stream>>>(W0, W1, W2, pos1, wb0, wb1, wb2, stats, out);
  layer0_kernel<<<M_/MT, 256, 0, stream>>>(pos1, pos2, f1, f2, idx, wb0, y0, stats);
  layerN_kernel<<<M_/MT, 256, 0, stream>>>(y0, wb1, stats,       g0, b0, y1, stats + 256, 128);
  layerN_kernel<<<M_/MT, 256, 0, stream>>>(y1, wb2, stats + 256, g1, b1, y2, stats + 512, 256);
  finalize_kernel<<<B_*N_, 256, 0, stream>>>(y2, stats + 512, g2, b2, out_feat);
}
